// Actor_44633300140735
// MI455X (gfx1250) — matrix operand order repaired and checked by reading, not compiled, not run on hardware
//
#include <hip/hip_runtime.h>

typedef __bf16 bf16;
typedef __attribute__((ext_vector_type(16))) __bf16 v16bf;
typedef __attribute__((ext_vector_type(8)))  float  v8f;
typedef __attribute__((ext_vector_type(4)))  unsigned int u32x4;
typedef __attribute__((ext_vector_type(8)))  int i32x8;
typedef __attribute__((ext_vector_type(4)))  int i32x4;

#if defined(__has_builtin)
# if __has_builtin(__builtin_amdgcn_tensor_load_to_lds) && __has_builtin(__builtin_amdgcn_s_wait_tensorcnt)
#  define USE_TDM 1
# endif
#endif
#ifndef USE_TDM
# define USE_TDM 0
#endif

struct F8 { unsigned int w[8]; };

__device__ __forceinline__ v16bf load_frag(const bf16* p0, const bf16* p1) {
  uint4 x = *reinterpret_cast<const uint4*>(p0);
  uint4 y = *reinterpret_cast<const uint4*>(p1);
  F8 r{{x.x, x.y, x.z, x.w, y.x, y.y, y.z, y.w}};
  return __builtin_bit_cast(v16bf, r);
}

__device__ __forceinline__ v8f wmma_bf16(v16bf a, v16bf b, v8f c) {
  return __builtin_amdgcn_wmma_f32_16x16x32_bf16(false, a, false, b, (short)0, c, false, false);
}

__device__ __forceinline__ bf16 f2bf(float f) {
  unsigned u = __builtin_bit_cast(unsigned, f);
  unsigned r = u + 0x7fffu + ((u >> 16) & 1u);
  unsigned short h = (unsigned short)(r >> 16);
  return __builtin_bit_cast(bf16, h);
}

// ---------------------------------------------------------------------------
// Generic WMMA GEMM: C[m,n] = alpha * sum_k W[m,k]*Xt[n,k] (+bias[m]) (+relu)
// (+residual). W: bf16 (M,K) row-major ld=ldw.  Xt: bf16 (N,K) row-major
// ld=ldx (i.e. X pre-transposed so B-fragments are contiguous 16B loads).
// Output row remap: row_out = m*crscale + z*crzoff (head interleaving).
// Batched over blockIdx.z with element strides sW/sX/sC.
// ---------------------------------------------------------------------------
__global__ __launch_bounds__(128)
void gemm_wmma_k(const bf16* W, const bf16* X, float* C,
                 const float* bias, const float* resid,
                 int M, int N, int K, int ldw, int ldx, int ldc,
                 long sW, long sX, long sC,
                 int crscale, int crzoff, float alpha, int relu, int bstrz)
{
  int z = blockIdx.z;
  W += (long)z * sW; X += (long)z * sX;
  float* Cz = C + (long)z * sC;
  const float* Rz = resid ? resid + (long)z * sC : nullptr;
  int lane = threadIdx.x & 31;
  int wave = threadIdx.x >> 5;
  int half = lane >> 4;
  int l15  = lane & 15;
  int mbase = blockIdx.x * 16;
  int nbase = blockIdx.y * 64 + wave * 16;
  const bf16* wr = W + (long)(mbase + l15) * ldw + half * 8;
  const bf16* xr = X + (long)(nbase + l15) * ldx + half * 8;
  v8f acc = {};
  for (int k0 = 0; k0 < K; k0 += 32) {
    v16bf a = load_frag(wr + k0, wr + k0 + 16);
    v16bf b = load_frag(xr + k0, xr + k0 + 16);
    acc = wmma_bf16(a, b, acc);
  }
  int col = nbase + l15;
#pragma unroll
  for (int r = 0; r < 8; ++r) {
    int m = mbase + r + half * 8;
    float v = alpha * acc[r];
    if (bias) v += bias[m + z * bstrz];
    if (relu) v = fmaxf(v, 0.f);
    long idx = ((long)m * crscale + (long)z * crzoff) * ldc + col;
    if (Rz) v += Rz[idx];
    Cz[idx] = v;
  }
}

// dst[c*ldd + dcoff + r] = bf16(src[(r*rscale + roff0 + z*rzoff)*lds + c])
__global__ void transpose_convert_k(const float* src, bf16* dst,
    int R, int C, int lds, int rscale, int roff0, int rzoff,
    int ldd, int dcoff, long sS, long sD)
{
  int z = blockIdx.z;
  src += (long)z * sS;
  dst += (long)z * sD;
  int roff = roff0 + z * rzoff;
  long total = (long)R * C;
  for (long t = (long)blockIdx.x * blockDim.x + threadIdx.x; t < total;
       t += (long)gridDim.x * blockDim.x) {
    int  r = (int)(t % R);
    long c = t / R;
    dst[c * ldd + dcoff + r] = f2bf(src[(long)(r * rscale + roff) * lds + c]);
  }
}

// dst[m*K + k] = bf16(src[(m*rscale + roff0 + z*rzoff)*lds + k + koff])
__global__ void convert_bf_k(const float* src, bf16* dst,
    int M, int K, int lds, int rscale, int roff0, int rzoff,
    int koff, long sS, long sD)
{
  int z = blockIdx.z;
  src += (long)z * sS;
  dst += (long)z * sD;
  int roff = roff0 + z * rzoff;
  long total = (long)M * K;
  for (long t = (long)blockIdx.x * blockDim.x + threadIdx.x; t < total;
       t += (long)gridDim.x * blockDim.x) {
    int  k = (int)(t % K);
    long m = t / K;
    dst[m * K + k] = f2bf(src[(m * rscale + roff) * (long)lds + k + koff]);
  }
}

__global__ void softmax_rows_k(const float* src, bf16* dst, int C) {
  long row = blockIdx.x;
  src += row * C; dst += row * C;
  __shared__ float red[256];
  int t = threadIdx.x;
  float m = -1e30f;
  for (int i = t; i < C; i += 256) m = fmaxf(m, src[i]);
  red[t] = m; __syncthreads();
  for (int s = 128; s > 0; s >>= 1) { if (t < s) red[t] = fmaxf(red[t], red[t + s]); __syncthreads(); }
  m = red[0]; __syncthreads();
  float sum = 0.f;
  for (int i = t; i < C; i += 256) sum += expf(src[i] - m);
  red[t] = sum; __syncthreads();
  for (int s = 128; s > 0; s >>= 1) { if (t < s) red[t] += red[t + s]; __syncthreads(); }
  float inv = 1.f / red[0];
  for (int i = t; i < C; i += 256) dst[i] = f2bf(expf(src[i] - m) * inv);
}

// ---------------------------------------------------------------------------
// Fused cross-attention scorer.  pre0[d,i,j] = Cd + P0[d,i] + P1k[d,j],
// pre1[d,j,i] = Cd + P1q[d,j] + P0k[d,i] with Cd = (Wd @ dist)[d, j*1024+i]
// computed on the fly via WMMA.  dist tile (64 pairs x 256 k, f32) is staged
// into LDS by the Tensor Data Mover (4 x 16KB 2D descriptors, wave 0,
// s_wait_tensorcnt), converted once to a [pair][k] bf16 LDS tile, then the
// WMMA K-loop runs barrier-free out of LDS.
// scores0[i,j] = sum_d w2[d]*relu(pre0)+b2 ; score1[j,i] likewise.
// Block = 256 threads (8 waves); 64 pairs x 256 d per block.
// ---------------------------------------------------------------------------
__global__ __launch_bounds__(256)
void cross_score_k(const float* __restrict__ dist, const bf16* __restrict__ Wd,
                   const float* __restrict__ P0,  const float* __restrict__ P1k,
                   const float* __restrict__ P1q, const float* __restrict__ P0k,
                   const float* __restrict__ w2,  const float* __restrict__ b2,
                   float* __restrict__ scores0,   float* __restrict__ score1)
{
  const int NP = 131072;
  __shared__ alignas(16) float ldsF[64 * 64];    // 16KB raw f32 chunk [k][pair]
  __shared__ alignas(16) bf16  ldsB[64 * 256];   // 32KB bf16 tile  [pair][k]
  __shared__ float s0red[64];
  __shared__ float s1red[64];
  int tid  = threadIdx.x;
  int lane = tid & 31, wave = tid >> 5;
  int half = lane >> 4, l15 = lane & 15;
  int pair0 = blockIdx.x * 64;
  if (tid < 64) { s0red[tid] = 0.f; s1red[tid] = 0.f; }

  // ---- stage dist tile: 4 chunks of 64 k-rows x 64 pairs (f32) -> bf16 LDS
  for (int c = 0; c < 4; ++c) {
    __syncthreads();                       // ldsF reuse + s0red init visibility
#if USE_TDM
    if (wave == 0) {
      unsigned long long ga =
          (unsigned long long)(uintptr_t)(dist + (long)c * 64 * NP + pair0);
      unsigned ldsaddr = (unsigned)(uintptr_t)(&ldsF[0]);
      u32x4 g0;
      g0.x = 1u;                                            // count=1 (valid)
      g0.y = ldsaddr;                                       // lds_addr
      g0.z = (unsigned)(ga & 0xffffffffull);                // global_addr lo
      g0.w = (unsigned)((ga >> 32) & 0x1ffffffull) | (2u << 30); // hi | type=2
      i32x8 g1;
      g1[0] = (int)(2u << 16);               // data_size = 4B
      g1[1] = 0;                             // tensor_dim0[15:0] (131072 lo16=0)
      g1[2] = (int)((131072u >> 16) | (64u << 16)); // dim0 hi16 | tensor_dim1=64
      g1[3] = (int)(64u << 16);              // dim1 hi16=0 | tile_dim0=64
      g1[4] = (int)64;                       // tile_dim1=64, tile_dim2=0
      g1[5] = (int)131072;                   // tensor_dim0_stride lo32
      g1[6] = 0;                             // stride hi16 | dim1_stride lo16
      g1[7] = 0;
      i32x4 zz = {0, 0, 0, 0};
#if __clang_major__ >= 23
      i32x8 z8 = {};
      __builtin_amdgcn_tensor_load_to_lds(g0, g1, zz, zz, z8, 0);
#else
      __builtin_amdgcn_tensor_load_to_lds(g0, g1, zz, zz, 0);
#endif
      __builtin_amdgcn_s_wait_tensorcnt(0);
    }
#else
    for (int idx = tid; idx < 64 * 64; idx += 256) {
      int k = idx >> 6, p = idx & 63;
      ldsF[idx] = dist[(long)(c * 64 + k) * NP + pair0 + p];
    }
#endif
    __syncthreads();
    for (int idx = tid; idx < 64 * 64; idx += 256) {
      int k = idx >> 6, p = idx & 63;
      ldsB[p * 256 + c * 64 + k] = f2bf(ldsF[idx]);
    }
  }
  __syncthreads();

  // ---- barrier-free WMMA K-loop out of LDS
  v8f acc[2][4] = {};
  int dbase = wave * 32;
  const bf16* wr0 = Wd + (long)(dbase + l15) * 256 + half * 8;
  const bf16* wr1 = Wd + (long)(dbase + 16 + l15) * 256 + half * 8;
  for (int k0 = 0; k0 < 256; k0 += 32) {
    v16bf a0 = load_frag(wr0 + k0, wr0 + k0 + 16);
    v16bf a1 = load_frag(wr1 + k0, wr1 + k0 + 16);
#pragma unroll
    for (int nt = 0; nt < 4; ++nt) {
      const bf16* lb = ldsB + (nt * 16 + l15) * 256 + k0 + half * 8;
      v16bf b = load_frag(lb, lb + 16);
      acc[0][nt] = wmma_bf16(a0, b, acc[0][nt]);
      acc[1][nt] = wmma_bf16(a1, b, acc[1][nt]);
    }
  }

  // ---- fused epilogue: relu + w2-dot for both directions, LDS reduction
  float bb = b2[0];
#pragma unroll
  for (int nt = 0; nt < 4; ++nt) {
    int pair = pair0 + nt * 16 + l15;
    int i = pair & 1023;
    int j = pair >> 10;
    float s0 = 0.f, s1 = 0.f;
#pragma unroll
    for (int mt = 0; mt < 2; ++mt) {
#pragma unroll
      for (int r = 0; r < 8; ++r) {
        int d = dbase + mt * 16 + r + half * 8;
        float c  = acc[mt][nt][r];
        float wd = w2[d];
        s0 += wd * fmaxf(c + P0[(long)d * 1024 + i] + P1k[d * 128 + j], 0.f);
        s1 += wd * fmaxf(c + P1q[d * 128 + j] + P0k[(long)d * 1024 + i], 0.f);
      }
    }
    atomicAdd(&s0red[nt * 16 + l15], s0);
    atomicAdd(&s1red[nt * 16 + l15], s1);
  }
  __syncthreads();
  if (tid < 64) {
    int pair = pair0 + tid;
    int i = pair & 1023;
    int j = pair >> 10;
    scores0[(long)i * 128 + j]  = s0red[tid] + bb;
    score1[(long)j * 1024 + i]  = s1red[tid] + bb;
  }
}

// log_softmax over the 128-dim (axis -2) of score1 (128 x 1024)
__global__ void logsoftmax_cols_k(const float* s, float* out) {
  int i = blockIdx.x;       // 0..1023
  int j = threadIdx.x;      // 0..127
  __shared__ float red[128];
  float v = s[(long)j * 1024 + i];
  red[j] = v; __syncthreads();
  for (int st = 64; st > 0; st >>= 1) { if (j < st) red[j] = fmaxf(red[j], red[j + st]); __syncthreads(); }
  float m = red[0]; __syncthreads();
  red[j] = expf(v - m); __syncthreads();
  for (int st = 64; st > 0; st >>= 1) { if (j < st) red[j] += red[j + st]; __syncthreads(); }
  out[(long)j * 1024 + i] = v - (m + logf(red[0]));
}

__global__ void build_Z_k(const float* ls, const float* alpha, float* Z) {
  float a = alpha[0];
  for (int t = blockIdx.x * 256 + threadIdx.x; t < 129 * 1025; t += gridDim.x * 256) {
    int r = t / 1025, c = t % 1025;
    Z[t] = (r < 128 && c < 1024) ? ls[r * 1024 + c] : a;
  }
}

__global__ void zero_k(float* p, int n) {
  int t = blockIdx.x * 256 + threadIdx.x;
  if (t < n) p[t] = 0.f;
}

__global__ void sink_row_k(const float* Z, const float* v, float* u) {
  int r = blockIdx.x;                     // 0..128
  const float* zr = Z + (long)r * 1025;
  __shared__ float red[256];
  int t = threadIdx.x;
  float m = -1e30f;
  for (int i = t; i < 1025; i += 256) m = fmaxf(m, zr[i] + v[i]);
  red[t] = m; __syncthreads();
  for (int s = 128; s > 0; s >>= 1) { if (t < s) red[t] = fmaxf(red[t], red[t + s]); __syncthreads(); }
  m = red[0]; __syncthreads();
  float sum = 0.f;
  for (int i = t; i < 1025; i += 256) sum += expf(zr[i] + v[i] - m);
  red[t] = sum; __syncthreads();
  for (int s = 128; s > 0; s >>= 1) { if (t < s) red[t] += red[t + s]; __syncthreads(); }
  if (t == 0) {
    const float norm = -logf(1152.f);
    float lm = (r < 128) ? norm : (logf(1024.f) + norm);
    u[r] = lm - (m + logf(red[0]));
  }
}

__global__ void sink_col_k(const float* Z, const float* u, float* v) {
  int c = blockIdx.x;                     // 0..1024
  int t = threadIdx.x;                    // 256 threads, 129 active rows
  __shared__ float red[256];
  float val = (t < 129) ? Z[(long)t * 1025 + c] + u[t] : -1e30f;
  red[t] = val; __syncthreads();
  for (int st = 128; st > 0; st >>= 1) { if (t < st) red[t] = fmaxf(red[t], red[t + st]); __syncthreads(); }
  float m = red[0]; __syncthreads();
  red[t] = (t < 129) ? expf(val - m) : 0.f; __syncthreads();
  for (int st = 128; st > 0; st >>= 1) { if (t < st) red[t] += red[t + st]; __syncthreads(); }
  if (t == 0) {
    const float norm = -logf(1152.f);
    float ln = (c < 1024) ? norm : (logf(128.f) + norm);
    v[c] = ln - (m + logf(red[0]));
  }
}

__global__ void finalZ_k(const float* Z, const float* u, const float* v,
                         float* zf, float* pmin, float* pmax) {
  __shared__ float rmn[256], rmx[256];
  int t = threadIdx.x;
  int g = blockIdx.x * 256 + t;           // grid = 512 -> 131072
  int j = g >> 10, i = g & 1023;
  const float norm = -logf(1152.f);
  float val = Z[(long)j * 1025 + i] + u[j] + v[i] - norm;
  zf[g] = val;
  rmn[t] = val; rmx[t] = val; __syncthreads();
  for (int s = 128; s > 0; s >>= 1) {
    if (t < s) { rmn[t] = fminf(rmn[t], rmn[t + s]); rmx[t] = fmaxf(rmx[t], rmx[t + s]); }
    __syncthreads();
  }
  if (t == 0) { pmin[blockIdx.x] = rmn[0]; pmax[blockIdx.x] = rmx[0]; }
}

__global__ void reduce_minmax_k(const float* pmin, const float* pmax, float* smm) {
  __shared__ float rmn[512], rmx[512];
  int t = threadIdx.x;
  rmn[t] = pmin[t]; rmx[t] = pmax[t]; __syncthreads();
  for (int s = 256; s > 0; s >>= 1) {
    if (t < s) { rmn[t] = fminf(rmn[t], rmn[t + s]); rmx[t] = fmaxf(rmx[t], rmx[t + s]); }
    __syncthreads();
  }
  if (t == 0) { smm[0] = rmn[0]; smm[1] = rmx[0]; }
}

__global__ void mask_out_k(const float* zf, const float* smm, const int* fidx,
                           const int* lmb, const unsigned char* unreach, float* out) {
  int g = blockIdx.x * 256 + threadIdx.x;
  int j = g >> 10, i = g & 1023;
  float sm = smm[0] - smm[1];
  int f0 = fidx[i * 2 + 0], f1 = fidx[i * 2 + 1];
  int lo0 = lmb[j * 4 + 0], hi0 = lmb[j * 4 + 1];
  int lo1 = lmb[j * 4 + 2], hi1 = lmb[j * 4 + 3];
  bool inval = (f0 < lo0) || (f0 >= hi0) || (f1 < lo1) || (f1 >= hi1);
  float v = zf[g];
  if (inval)      v += sm - 40.f;
  if (unreach[g]) v += sm - 20.f;
  out[g] = v * 15.f;
}

// ---------------------------------------------------------------------------
extern "C" void kernel_launch(void* const* d_in, const int* in_sizes, int n_in,
                              void* d_out, int out_size, void* d_ws, size_t ws_size,
                              hipStream_t stream)
{
  (void)in_sizes; (void)out_size; (void)ws_size;
  if (n_in < 43) return;
  const float* in_d0  = (const float*)d_in[0];
  const float* in_d1  = (const float*)d_in[1];
  const int*   lmb    = (const int*)d_in[4];
  const int*   fidx   = (const int*)d_in[5];
  const float* dist   = (const float*)d_in[8];
  const unsigned char* unreach = (const unsigned char*)d_in[9];
  const float* bin_score = (const float*)d_in[10];
  const float* L[4][8];
  { int ix = 11;
    for (int l = 0; l < 4; ++l)
      for (int p = 0; p < 8; ++p) L[l][p] = (const float*)d_in[ix++];
  }

  char* base = (char*)d_ws;
  size_t off = 0;
  auto alloc = [&](size_t bytes) -> void* {
    off = (off + 255) & ~(size_t)255;
    void* p = base + off;
    off += bytes;
    return p;
  };
  float* d0f  = (float*)alloc((size_t)256*1024*4);
  float* d1f  = (float*)alloc((size_t)256*128*4);
  bf16* WPROJ = (bf16*)alloc((size_t)3*256*256*2);
  bf16* WA    = (bf16*)alloc((size_t)512*512*2);
  bf16* WB    = (bf16*)alloc((size_t)256*512*2);
  bf16* WM    = (bf16*)alloc((size_t)256*256*2);
  bf16* WQb   = (bf16*)alloc((size_t)256*256*2);
  bf16* WKb   = (bf16*)alloc((size_t)256*256*2);
  bf16* WDb   = (bf16*)alloc((size_t)256*256*2);
  bf16* XT1b  = (bf16*)alloc((size_t)1024*512*2);
  bf16* XT2b  = (bf16*)alloc((size_t)1024*512*2);
  bf16* D0T   = (bf16*)alloc((size_t)1024*256*2);
  bf16* D1T   = (bf16*)alloc((size_t)128*256*2);
  bf16* D0B   = (bf16*)alloc((size_t)256*1024*2);
  bf16* D1B   = (bf16*)alloc((size_t)256*128*2);
  float* QKV  = (float*)alloc((size_t)3*256*1024*4);
  bf16* QHT   = (bf16*)alloc((size_t)4*1024*64*2);
  bf16* KHT   = (bf16*)alloc((size_t)4*1024*64*2);
  bf16* VHB   = (bf16*)alloc((size_t)4*64*1024*2);
  float* Sb   = (float*)alloc((size_t)4*1024*1024*4);
  bf16*  Pbf  = (bf16*)alloc((size_t)4*1024*1024*2);
  float* MSG  = (float*)alloc((size_t)256*1024*4);
  float* MSG1 = (float*)alloc((size_t)256*128*4);
  float* XMSG = (float*)alloc((size_t)256*1024*4);
  float* H1   = (float*)alloc((size_t)512*1024*4);
  float* P0f  = (float*)alloc((size_t)256*1024*4);
  float* P0Kf = (float*)alloc((size_t)256*1024*4);
  float* P1Qf = (float*)alloc((size_t)256*128*4);
  float* P1Kf = (float*)alloc((size_t)256*128*4);
  float* SC0  = (float*)alloc((size_t)1024*128*4);
  float* SC1  = (float*)alloc((size_t)128*1024*4);
  bf16*  PR0  = (bf16*)alloc((size_t)1024*128*2);
  bf16*  PR1  = (bf16*)alloc((size_t)128*1024*2);
  float* LSB  = (float*)alloc((size_t)128*1024*4);
  float* Zb   = (float*)alloc((size_t)129*1025*4);
  float* Ub   = (float*)alloc((size_t)256*4);
  float* Vb   = (float*)alloc((size_t)1056*4);
  float* ZF   = (float*)alloc((size_t)131072*4);
  float* PMIN = (float*)alloc((size_t)512*4);
  float* PMAX = (float*)alloc((size_t)512*4);
  float* SMM  = (float*)alloc((size_t)2*4);

  hipMemcpyAsync(d0f, in_d0, (size_t)256*1024*4, hipMemcpyDeviceToDevice, stream);
  hipMemcpyAsync(d1f, in_d1, (size_t)256*128*4,  hipMemcpyDeviceToDevice, stream);

  auto gridFor = [](long total) -> unsigned {
    long b = (total + 255) / 256;
    if (b > 4096) b = 4096;
    if (b < 1) b = 1;
    return (unsigned)b;
  };
  auto tconv = [&](const float* src, bf16* dst, int R, int C, int lds,
                   int rscale, int roff0, int rzoff, int ldd, int dcoff,
                   long sS, long sD, int nz) {
    transpose_convert_k<<<dim3(gridFor((long)R * C), 1, nz), 256, 0, stream>>>(
        src, dst, R, C, lds, rscale, roff0, rzoff, ldd, dcoff, sS, sD);
  };
  auto conv = [&](const float* src, bf16* dst, int M, int K, int lds,
                  int rscale, int roff0, int rzoff, int koff, long sS, long sD, int nz) {
    convert_bf_k<<<dim3(gridFor((long)M * K), 1, nz), 256, 0, stream>>>(
        src, dst, M, K, lds, rscale, roff0, rzoff, koff, sS, sD);
  };
  auto gemm = [&](const bf16* W, const bf16* X, float* C, const float* bias,
                  const float* resid, int M, int N, int K, int ldw, int ldx, int ldc,
                  long sW, long sX, long sC, int crscale, int crzoff,
                  float alpha, int relu, int bstrz, int nz) {
    gemm_wmma_k<<<dim3(M / 16, N / 64, nz), 128, 0, stream>>>(
        W, X, C, bias, resid, M, N, K, ldw, ldx, ldc, sW, sX, sC,
        crscale, crzoff, alpha, relu, bstrz);
  };
  auto mlp_block = [&](float* dptr, const float* msgptr, int n,
                       const float* w1, const float* b1,
                       const float* w2p, const float* b2p) {
    tconv(dptr,   XT1b, 256, n, n, 1, 0, 0, 512, 0,   0, 0, 1);
    tconv(msgptr, XT1b, 256, n, n, 1, 0, 0, 512, 256, 0, 0, 1);
    conv(w1, WA, 512, 512, 512, 1, 0, 0, 0, 0, 0, 1);
    gemm(WA, XT1b, H1, b1, nullptr, 512, n, 512, 512, 512, n, 0, 0, 0, 1, 0, 1.f, 1, 0, 1);
    tconv(H1, XT2b, 512, n, n, 1, 0, 0, 512, 0, 0, 0, 1);
    conv(w2p, WB, 256, 512, 512, 1, 0, 0, 0, 0, 0, 1);
    gemm(WB, XT2b, dptr, b2p, dptr, 256, n, 512, 512, 512, n, 0, 0, 0, 1, 0, 1.f, 0, 0, 1);
  };

  auto self_layer = [&](float* dptr, int n, const float* const* Lp) {
    const float *merge_b = Lp[0], *merge_w = Lp[1], *mlp1_b = Lp[2], *mlp1_w = Lp[3];
    const float *mlp2_b = Lp[4], *mlp2_w = Lp[5], *proj_b = Lp[6], *proj_w = Lp[7];
    tconv(dptr, XT1b, 256, n, n, 1, 0, 0, 256, 0, 0, 0, 1);
    conv(proj_w, WPROJ, 256, 256, 256, 1, 0, 0, 0, 65536, 65536, 3);
    gemm(WPROJ, XT1b, QKV, proj_b, nullptr, 256, n, 256, 256, 256, n,
         65536, 0, 256L * n, 1, 0, 1.f, 0, 256, 3);
    float* Q  = QKV;
    float* Kf = QKV + 256L * n;
    float* Vf = QKV + 512L * n;
    // per-head strided transposes (head h uses rows 4a+h)
    tconv(Q,  QHT, 64, n, n, 4, 0, 1, 64, 0, 0, 64L * n, 4);
    tconv(Kf, KHT, 64, n, n, 4, 0, 1, 64, 0, 0, 64L * n, 4);
    conv(Vf, VHB, 64, n, n, 4, 0, 1, 0, 0, 64L * n, 4);
    gemm(QHT, KHT, Sb, nullptr, nullptr, n, n, 64, 64, 64, n,
         64L * n, 64L * n, (long)n * n, 1, 0, 0.125f, 0, 0, 4);
    softmax_rows_k<<<4 * n, 256, 0, stream>>>(Sb, Pbf, n);
    gemm(VHB, Pbf, MSG, nullptr, nullptr, 64, n, n, n, n, n,
         64L * n, (long)n * n, 0, 4, 1, 1.f, 0, 0, 4);   // rows interleave 4a+h
    tconv(MSG, XT2b, 256, n, n, 1, 0, 0, 256, 0, 0, 0, 1);
    conv(merge_w, WM, 256, 256, 256, 1, 0, 0, 0, 0, 0, 1);
    gemm(WM, XT2b, XMSG, merge_b, nullptr, 256, n, 256, 256, 256, n,
         0, 0, 0, 1, 0, 1.f, 0, 0, 1);
    mlp_block(dptr, XMSG, n, mlp1_w, mlp1_b, mlp2_w, mlp2_b);
  };

  auto cross_layer = [&](const float* const* Lp) {
    const float *a1b = Lp[0], *a1w = Lp[1], *a2b = Lp[2], *a2w = Lp[3];
    const float *m1b = Lp[4], *m1w = Lp[5], *m2b = Lp[6], *m2w = Lp[7];
    tconv(d0f, D0T, 256, 1024, 1024, 1, 0, 0, 256, 0, 0, 0, 1);
    tconv(d1f, D1T, 256, 128, 128, 1, 0, 0, 256, 0, 0, 0, 1);
    conv(a1w, WQb, 256, 256, 768, 1, 0, 0, 0,   0, 0, 1);
    conv(a1w, WKb, 256, 256, 768, 1, 0, 0, 256, 0, 0, 1);
    conv(a1w, WDb, 256, 256, 768, 1, 0, 0, 512, 0, 0, 1);
    gemm(WQb, D0T, P0f,  a1b, nullptr, 256, 1024, 256, 256, 256, 1024, 0,0,0,1,0,1.f,0,0,1);
    gemm(WQb, D1T, P1Qf, a1b, nullptr, 256, 128,  256, 256, 256, 128,  0,0,0,1,0,1.f,0,0,1);
    gemm(WKb, D0T, P0Kf, nullptr, nullptr, 256, 1024, 256, 256, 256, 1024, 0,0,0,1,0,1.f,0,0,1);
    gemm(WKb, D1T, P1Kf, nullptr, nullptr, 256, 128,  256, 256, 256, 128,  0,0,0,1,0,1.f,0,0,1);
    cross_score_k<<<2048, 256, 0, stream>>>(dist, WDb, P0f, P1Kf, P1Qf, P0Kf,
                                            a2w, a2b, SC0, SC1);
    softmax_rows_k<<<1024, 256, 0, stream>>>(SC0, PR0, 128);
    softmax_rows_k<<<128,  256, 0, stream>>>(SC1, PR1, 1024);
    conv(d1f, D1B, 256, 128, 128, 1, 0, 0, 0, 0, 0, 1);
    conv(d0f, D0B, 256, 1024, 1024, 1, 0, 0, 0, 0, 0, 1);
    gemm(D1B, PR0, MSG,  nullptr, nullptr, 256, 1024, 128, 128, 128, 1024, 0,0,0,1,0,1.f,0,0,1);
    gemm(D0B, PR1, MSG1, nullptr, nullptr, 256, 128, 1024, 1024, 1024, 128, 0,0,0,1,0,1.f,0,0,1);
    mlp_block(d0f, MSG,  1024, m1w, m1b, m2w, m2b);
    mlp_block(d1f, MSG1, 128,  m1w, m1b, m2w, m2b);
  };

  // layer order: self, cross, self, cross
  self_layer(d0f, 1024, L[0]);
  self_layer(d1f, 128,  L[0]);
  cross_layer(L[1]);
  self_layer(d0f, 1024, L[2]);
  self_layer(d1f, 128,  L[2]);
  cross_layer(L[3]);

  // final: log_softmax(axis=-2) -> Sinkhorn -> mask
  logsoftmax_cols_k<<<1024, 128, 0, stream>>>(SC1, LSB);
  build_Z_k<<<gridFor(129L * 1025), 256, 0, stream>>>(LSB, bin_score, Zb);
  zero_k<<<2, 256, 0, stream>>>(Ub, 129);
  zero_k<<<5, 256, 0, stream>>>(Vb, 1025);
  for (int it = 0; it < 5; ++it) {
    sink_row_k<<<129, 256, 0, stream>>>(Zb, Vb, Ub);
    sink_col_k<<<1025, 256, 0, stream>>>(Zb, Ub, Vb);
  }
  finalZ_k<<<512, 256, 0, stream>>>(Zb, Ub, Vb, ZF, PMIN, PMAX);
  reduce_minmax_k<<<1, 512, 0, stream>>>(PMIN, PMAX, SMM);
  mask_out_k<<<512, 256, 0, stream>>>(ZF, SMM, fidx, lmb, unreach, (float*)d_out);
}
